// SpectralLanguageModel_37469294690526
// MI455X (gfx1250) — matrix operand order repaired and checked by reading, not compiled, not run on hardware
//
#include <hip/hip_runtime.h>
#include <hip/hip_bf16.h>
#include <math.h>

// ---------------- model dims ----------------
#define MV   50257
#define ME   768
#define MH   1536
#define ML   4
#define MNH  12
#define MHD  128
#define MB   2
#define MS   2048
#define MT   (MB*MS)      // 4096 tokens
#define MF   (MS/2+1)     // 1025 rfft bins
#define MTOPK 102         // int(1025*0.1)
#define MFFN (4*MH)       // 6144
#define MAXK 6144         // largest GEMM K (for zero pad row)

typedef __attribute__((ext_vector_type(16))) __bf16 v16bf;
typedef __attribute__((ext_vector_type(8)))  float  v8f;

__device__ __forceinline__ unsigned short f2bf(float f) {
  unsigned int u = __float_as_uint(f);
  u += 0x7fffu + ((u >> 16) & 1u);
  return (unsigned short)(u >> 16);
}
__device__ __forceinline__ float sigmoid_f(float x) { return 1.0f / (1.0f + expf(-x)); }

// ---------------------------------------------------------------------------
// WMMA GEMM: out = epilogue(A[M,K] * B[N,K]^T + bias, res)
// A, B bf16 (u16 storage), f32 accum via v_wmma_f32_16x16x32_bf16.
// Block: 256 thr = 8 waves; block tile 128(M) x 128(N); wave tile 32x64
// (2x4 wmma tiles -> 8 wmma per 12 b128 loads per K-step of 32).
// OOB N columns read from a zeroed pad row => no per-iteration masking.
// ---------------------------------------------------------------------------
constexpr int EP_F32       = 0;  // f32 out = acc + bias
constexpr int EP_F32_RES   = 1;  // f32 out = res + acc + bias
constexpr int EP_BF16      = 2;  // bf16 out = acc + bias
constexpr int EP_BF16_GELU = 3;  // bf16 out = gelu(acc + bias)  (exact erf gelu)

template<int EPI>
__global__ __launch_bounds__(256) void wmma_gemm_kernel(
    const unsigned short* __restrict__ A,    // bf16 [M,K] row-major
    const unsigned short* __restrict__ Bw,   // bf16 [N,K] row-major (weights)
    const unsigned short* __restrict__ zpad, // zeroed row, >= K elements
    const float* __restrict__ bias,          // [N] or nullptr
    const float* __restrict__ res,           // [M,N] or nullptr
    void* __restrict__ outp,                 // float* or bf16(u16)*
    int M, int N, int K)
{
  const int lane = threadIdx.x & 31;
  const int wave = threadIdx.x >> 5;
  const int l16  = lane & 15;
  const int lh   = lane >> 4;                // half-wave id

  const int tile_m = blockIdx.y * 128 + (wave >> 1) * 32;
  const int tile_n = blockIdx.x * 128 + (wave & 1)  * 64;

  int ar0 = tile_m + l16;        if (ar0 > M - 1) ar0 = M - 1;
  int ar1 = tile_m + 16 + l16;   if (ar1 > M - 1) ar1 = M - 1;

  // A 16-bit 16x32 layout: lane<16 -> K {0..7, 16..23}; lane>=16 -> +8
  const unsigned short* pa0 = A + (size_t)ar0 * K + lh * 8;
  const unsigned short* pa1 = A + (size_t)ar1 * K + lh * 8;
  // B 16-bit 32x16 layout: lane<16 -> K {0..7, 16..23}; lane>=16 -> +8
  const unsigned short* pb[4];
#pragma unroll
  for (int j = 0; j < 4; ++j) {
    const int c = tile_n + j * 16 + l16;
    pb[j] = (c < N) ? (Bw + (size_t)c * K + lh * 8) : (zpad + lh * 8);
  }

  v8f acc[2][4];
#pragma unroll
  for (int i = 0; i < 2; ++i)
#pragma unroll
    for (int j = 0; j < 4; ++j)
      acc[i][j] = (v8f){0.f,0.f,0.f,0.f,0.f,0.f,0.f,0.f};

  union Frag { v16bf v; uint4 u[2]; };

  for (int k = 0; k < K; k += 32) {
    if (k + 128 < K) {                       // uniform condition
      __builtin_prefetch(pa0 + k + 128, 0, 3);
      __builtin_prefetch(pb[0] + k + 128, 0, 3);
    }
    Frag fa0, fa1, fb[4];
    fa0.u[0] = *(const uint4*)(pa0 + k);
    fa0.u[1] = *(const uint4*)(pa0 + k + 16);
    fa1.u[0] = *(const uint4*)(pa1 + k);
    fa1.u[1] = *(const uint4*)(pa1 + k + 16);
#pragma unroll
    for (int j = 0; j < 4; ++j) {
      fb[j].u[0] = *(const uint4*)(pb[j] + k);
      fb[j].u[1] = *(const uint4*)(pb[j] + k + 16);
    }
#pragma unroll
    for (int j = 0; j < 4; ++j) {
      acc[0][j] = __builtin_amdgcn_wmma_f32_16x16x32_bf16(false, fa0.v, false, fb[j].v, (short)0, acc[0][j], false, false);
      acc[1][j] = __builtin_amdgcn_wmma_f32_16x16x32_bf16(false, fa1.v, false, fb[j].v, (short)0, acc[1][j], false, false);
    }
  }

  float* outF = (float*)outp;
  unsigned short* outB = (unsigned short*)outp;
  const int rb_in = 8 * lh;                  // D layout: VGPR r, half h -> M = r + 8h
#pragma unroll
  for (int i = 0; i < 2; ++i) {
    const int rowb = tile_m + i * 16 + rb_in;
#pragma unroll
    for (int j = 0; j < 4; ++j) {
      const int col = tile_n + j * 16 + l16;
      if (col >= N) continue;
      const float bv = bias ? bias[col] : 0.0f;
      v8f a = acc[i][j];
#pragma unroll
      for (int r = 0; r < 8; ++r) {
        const int row = rowb + r;
        if (row >= M) continue;
        const size_t idx = (size_t)row * N + col;
        float v = a[r] + bv;
        if (EPI == EP_F32)           outF[idx] = v;
        else if (EPI == EP_F32_RES)  outF[idx] = res[idx] + v;
        else if (EPI == EP_BF16)     outB[idx] = f2bf(v);
        else {
          float g = 0.5f * v * (1.0f + erff(v * 0.70710678118654752f));
          outB[idx] = f2bf(g);
        }
      }
    }
  }
}

// ---------------------------------------------------------------------------
// Stockham radix-2 complex FFT of length 2048 in LDS (forward, e^{-2pi i}).
// ---------------------------------------------------------------------------
__device__ float2* fft2048_stages(float2* src, float2* dst) {
  int n = 2048, ls = 0;
  while (n > 1) {
    const int m = n >> 1;
    for (int t = threadIdx.x; t < 1024; t += 256) {
      const int q = t & ((1 << ls) - 1);
      const int p = t >> ls;
      const float ang = -6.283185307179586f * (float)p / (float)n;
      const float cw = cosf(ang), sw = sinf(ang);
      const float2 a = src[q + (p << ls)];
      const float2 b = src[q + ((p + m) << ls)];
      dst[q + ((2 * p) << ls)] = make_float2(a.x + b.x, a.y + b.y);
      const float dx = a.x - b.x, dy = a.y - b.y;
      dst[q + ((2 * p + 1) << ls)] = make_float2(dx * cw - dy * sw, dx * sw + dy * cw);
    }
    __syncthreads();
    float2* tt = src; src = dst; dst = tt;
    n = m; ++ls;
  }
  return src;
}

// rfft over sequence axis, norm='ortho'. xn: [B,S,H] f32 -> Xc: [B,H,F] complex
__global__ __launch_bounds__(256) void rfft_kernel(const float* __restrict__ xn,
                                                   float2* __restrict__ Xc) {
  __shared__ float2 sA[2048];
  __shared__ float2 sB[2048];
  const int b  = blockIdx.x / MH;
  const int ch = blockIdx.x % MH;
  for (int s = threadIdx.x; s < MS; s += 256)
    sA[s] = make_float2(xn[((size_t)b * MS + s) * MH + ch], 0.f);
  __syncthreads();
  float2* r = fft2048_stages(sA, sB);
  const float sc = 0.022097086912079608f;   // 1/sqrt(2048)
  float2* o = Xc + ((size_t)b * MH + ch) * MF;
  for (int f = threadIdx.x; f < MF; f += 256)
    o[f] = make_float2(r[f].x * sc, r[f].y * sc);
}

// irfft (ortho) with fused mask * sigmoid(hw); output bf16 y: [B,S,H]
__global__ __launch_bounds__(256) void irfft_kernel(const float2* __restrict__ Xc,
                                                    const float* __restrict__ mask, // [B,NH,F]
                                                    const float* __restrict__ hw,   // [NH,HD]
                                                    unsigned short* __restrict__ y) {
  __shared__ float2 sA[2048];
  __shared__ float2 sB[2048];
  const int b  = blockIdx.x / MH;
  const int ch = blockIdx.x % MH;
  const int nh = ch >> 7, hd = ch & 127;
  const float w = sigmoid_f(hw[nh * MHD + hd]);
  const float2* xi = Xc + ((size_t)b * MH + ch) * MF;
  const float*  mk = mask + ((size_t)b * MNH + nh) * MF;
  for (int f = threadIdx.x; f < MF; f += 256) {
    float2 v = xi[f];
    const float mv = mk[f] * w;
    v.x *= mv; v.y *= mv;
    sA[f] = make_float2(v.x, -v.y);                 // conj(Xfull[f])
    if (f >= 1 && f <= MS / 2 - 1) sA[MS - f] = v;  // Hermitian mirror (conj of conj)
  }
  __syncthreads();
  float2* r = fft2048_stages(sA, sB);
  const float sc = 0.022097086912079608f;   // 1/sqrt(2048)
  for (int s = threadIdx.x; s < MS; s += 256)
    y[((size_t)b * MS + s) * MH + ch] = f2bf(r[s].x * sc);
}

// scores[b,nh,f] = mean_hd( |X[b,nh*HD+hd,f]| * sigmoid(fi[nh,hd]) )
__global__ __launch_bounds__(256) void score_kernel(const float2* __restrict__ Xc,
                                                    const float* __restrict__ fi,
                                                    float* __restrict__ scores) {
  const int b = blockIdx.x / MNH, nh = blockIdx.x % MNH;
  __shared__ float imp[MHD];
  for (int i = threadIdx.x; i < MHD; i += 256) imp[i] = sigmoid_f(fi[nh * MHD + i]);
  __syncthreads();
  const float2* base = Xc + ((size_t)b * MH + nh * MHD) * MF;
  float* so = scores + ((size_t)b * MNH + nh) * MF;
  for (int f = threadIdx.x; f < MF; f += 256) {
    float s = 0.f;
    for (int hd = 0; hd < MHD; ++hd) {
      const float2 v = base[(size_t)hd * MF + f];
      s += sqrtf(v.x * v.x + v.y * v.y) * imp[hd];
    }
    so[f] = s * (1.0f / (float)MHD);
  }
}

// top-k(102) over F per (b,nh): iterative argmax (lowest index wins ties) -> 0/1 mask
__global__ __launch_bounds__(256) void topk_kernel(const float* __restrict__ scores,
                                                   float* __restrict__ mask) {
  __shared__ float sc[MF];
  __shared__ float rv[256];
  __shared__ int   ri[256];
  const float* s = scores + (size_t)blockIdx.x * MF;
  float* m = mask + (size_t)blockIdx.x * MF;
  for (int f = threadIdx.x; f < MF; f += 256) { sc[f] = s[f]; m[f] = 0.0f; }
  __syncthreads();
  for (int it = 0; it < MTOPK; ++it) {
    float best = -3.4e38f; int bi = MF;
    for (int f = threadIdx.x; f < MF; f += 256) {
      const float v = sc[f];
      if (v > best) { best = v; bi = f; }
    }
    rv[threadIdx.x] = best; ri[threadIdx.x] = bi;
    __syncthreads();
    for (int o = 128; o > 0; o >>= 1) {
      if (threadIdx.x < o) {
        const float ov = rv[threadIdx.x + o]; const int oi = ri[threadIdx.x + o];
        if (ov > rv[threadIdx.x] || (ov == rv[threadIdx.x] && oi < ri[threadIdx.x])) {
          rv[threadIdx.x] = ov; ri[threadIdx.x] = oi;
        }
      }
      __syncthreads();
    }
    if (threadIdx.x == 0) { m[ri[0]] = 1.0f; sc[ri[0]] = -3.4e38f; }
    __syncthreads();
  }
}

// LayerNorm per row (D elements); OUT_BF16 selects output format
template<int OUT_BF16>
__global__ __launch_bounds__(256) void layernorm_kernel(const float* __restrict__ x,
                                                        const float* __restrict__ g,
                                                        const float* __restrict__ b,
                                                        int D,
                                                        float* __restrict__ outf,
                                                        unsigned short* __restrict__ outb) {
  const int row = blockIdx.x;
  const float* xr = x + (size_t)row * D;
  __shared__ float red[256];
  float s = 0.f;
  for (int i = threadIdx.x; i < D; i += 256) s += xr[i];
  red[threadIdx.x] = s; __syncthreads();
  for (int o = 128; o > 0; o >>= 1) { if (threadIdx.x < o) red[threadIdx.x] += red[threadIdx.x + o]; __syncthreads(); }
  const float mean = red[0] / (float)D;
  __syncthreads();
  s = 0.f;
  for (int i = threadIdx.x; i < D; i += 256) { const float d = xr[i] - mean; s += d * d; }
  red[threadIdx.x] = s; __syncthreads();
  for (int o = 128; o > 0; o >>= 1) { if (threadIdx.x < o) red[threadIdx.x] += red[threadIdx.x + o]; __syncthreads(); }
  const float inv = rsqrtf(red[0] / (float)D + 1e-5f);
  for (int i = threadIdx.x; i < D; i += 256) {
    const float v = (xr[i] - mean) * inv * g[i] + b[i];
    if (OUT_BF16) outb[(size_t)row * D + i] = f2bf(v);
    else          outf[(size_t)row * D + i] = v;
  }
}

// Embedding gather + RoPE -> bf16 [T, E]
__global__ __launch_bounds__(256) void embed_rope_kernel(const int* __restrict__ ids,
                                                         const float* __restrict__ emb,
                                                         unsigned short* __restrict__ x0) {
  const int tok = blockIdx.x;           // b*S + s
  const int pos = tok & (MS - 1);
  const float* e = emb + (size_t)ids[tok] * ME;
  unsigned short* o = x0 + (size_t)tok * ME;
  const float ln1e4 = 9.210340371976184f;   // ln(10000)
  for (int i = threadIdx.x; i < ME / 2; i += 256) {
    const float x1 = e[2 * i], x2 = e[2 * i + 1];
    const int j1 = (2 * i) % (ME / 2);      // index into emb vector (pre-rotation)
    const int j2 = (2 * i + 1) % (ME / 2);
    // inv_freq applied at emb position j: 10000^{-2*floor-free j/ME*2}; angle = pos * 10000^{-2 j / (2*ME/2) } ...
    const float f1 = expf(-((float)j1 / (float)ME) * ln1e4 * 2.0f);
    const float f2 = expf(-((float)j2 / (float)ME) * ln1e4 * 2.0f);
    const float t1 = (float)pos * f1;
    const float t2 = (float)pos * f2;
    const float o1 = x1 * cosf(t1) - x2 * sinf(t2);
    const float o2 = x1 * sinf(t1) + x2 * cosf(t2);
    o[i]          = f2bf(o1);
    o[ME / 2 + i] = f2bf(o2);
  }
}

// f32 -> bf16 convert, 4 elements/thread (n must be multiple of 4)
__global__ __launch_bounds__(256) void cvt_bf16x4_kernel(const float* __restrict__ in,
                                                         unsigned short* __restrict__ o,
                                                         size_t n4) {
  const size_t i = (size_t)blockIdx.x * 256 + threadIdx.x;
  if (i < n4) {
    const float4 v = ((const float4*)in)[i];
    ushort4 r;
    r.x = f2bf(v.x); r.y = f2bf(v.y); r.z = f2bf(v.z); r.w = f2bf(v.w);
    ((ushort4*)o)[i] = r;
  }
}

// zero-fill u16 buffer
__global__ __launch_bounds__(256) void zero_u16_kernel(unsigned short* __restrict__ p, int n) {
  for (int i = threadIdx.x + blockIdx.x * 256; i < n; i += gridDim.x * 256) p[i] = 0;
}

// ---------------------------------------------------------------------------
extern "C" void kernel_launch(void* const* d_in, const int* in_sizes, int n_in,
                              void* d_out, int out_size, void* d_ws, size_t ws_size,
                              hipStream_t stream) {
  (void)in_sizes; (void)n_in; (void)out_size; (void)ws_size;
  const int*   ids  = (const int*)  d_in[0];
  const float* emb  = (const float*)d_in[1];
  const float* in_w = (const float*)d_in[2];
  const float* in_b = (const float*)d_in[3];
  const float* lng  = (const float*)d_in[4];
  const float* lnb  = (const float*)d_in[5];
  const float* fi   = (const float*)d_in[6];
  const float* hw   = (const float*)d_in[7];
  const float* ow   = (const float*)d_in[8];
  const float* ob   = (const float*)d_in[9];
  const float* fng  = (const float*)d_in[10];
  const float* fnb  = (const float*)d_in[11];
  const float* fc1w = (const float*)d_in[12];
  const float* fc1b = (const float*)d_in[13];
  const float* fc2w = (const float*)d_in[14];
  const float* fc2b = (const float*)d_in[15];
  const float* ong  = (const float*)d_in[16];
  const float* onb  = (const float*)d_in[17];
  const float* opw  = (const float*)d_in[18];
  const float* opb  = (const float*)d_in[19];
  float* out = (float*)d_out;

  char* wsb = (char*)d_ws;
  size_t off = 0;
  auto take = [&](size_t bytes) -> char* {
    char* p = wsb + off;
    off = (off + bytes + 255) & ~(size_t)255;
    return p;
  };
  float*          Xact  = (float*)         take((size_t)MT * MH * 4);
  float*          XN    = (float*)         take((size_t)MT * MH * 4);
  float2*         FREQ  = (float2*)        take((size_t)MB * MH * MF * 8);
  unsigned short* BF1   = (unsigned short*)take((size_t)MT * MH * 2);
  unsigned short* HBF   = (unsigned short*)take((size_t)MT * MFFN * 2);
  unsigned short* BF2   = (unsigned short*)take((size_t)MT * ME * 2);
  unsigned short* WBUF  = (unsigned short*)take((size_t)MFFN * MH * 2);
  unsigned short* EMBB  = (unsigned short*)take((size_t)MV * ME * 2);
  unsigned short* ZPAD  = (unsigned short*)take((size_t)MAXK * 2);
  float*          SCOR  = (float*)         take((size_t)MB * MNH * MF * 4);
  float*          MASKB = (float*)         take((size_t)MB * MNH * MF * 4);

  auto cvt = [&](const float* src, unsigned short* dst, size_t n) {
    const size_t n4 = n / 4;
    const int blocks = (int)((n4 + 255) / 256);
    cvt_bf16x4_kernel<<<blocks, 256, 0, stream>>>(src, dst, n4);
  };
  auto gemm = [&](int epi, const unsigned short* A, const unsigned short* Bw,
                  const float* bias, const float* res, void* o, int M, int N, int K) {
    dim3 g((unsigned)((N + 127) / 128), (unsigned)((M + 127) / 128), 1);
    switch (epi) {
      case EP_F32:       wmma_gemm_kernel<EP_F32      ><<<g, 256, 0, stream>>>(A, Bw, ZPAD, bias, res, o, M, N, K); break;
      case EP_F32_RES:   wmma_gemm_kernel<EP_F32_RES  ><<<g, 256, 0, stream>>>(A, Bw, ZPAD, bias, res, o, M, N, K); break;
      case EP_BF16:      wmma_gemm_kernel<EP_BF16     ><<<g, 256, 0, stream>>>(A, Bw, ZPAD, bias, res, o, M, N, K); break;
      default:           wmma_gemm_kernel<EP_BF16_GELU><<<g, 256, 0, stream>>>(A, Bw, ZPAD, bias, res, o, M, N, K); break;
    }
  };

  // ---- prologue: zero pad row, emb->bf16, gather+rope, input projection ----
  zero_u16_kernel<<<8, 256, 0, stream>>>(ZPAD, MAXK);
  cvt(emb, EMBB, (size_t)MV * ME);
  embed_rope_kernel<<<MT, 256, 0, stream>>>(ids, emb, BF1);
  cvt(in_w, WBUF, (size_t)MH * ME);
  gemm(EP_F32, BF1, WBUF, in_b, nullptr, Xact, MT, MH, ME);

  // ---- layers ----
  for (int l = 0; l < ML; ++l) {
    const float* lng_l  = lng  + (size_t)l * MH;
    const float* lnb_l  = lnb  + (size_t)l * MH;
    const float* fi_l   = fi   + (size_t)l * MNH * MHD;
    const float* hw_l   = hw   + (size_t)l * MNH * MHD;
    const float* ow_l   = ow   + (size_t)l * MH * MH;
    const float* ob_l   = ob   + (size_t)l * MH;
    const float* fng_l  = fng  + (size_t)l * MH;
    const float* fnb_l  = fnb  + (size_t)l * MH;
    const float* fc1w_l = fc1w + (size_t)l * MFFN * MH;
    const float* fc1b_l = fc1b + (size_t)l * MFFN;
    const float* fc2w_l = fc2w + (size_t)l * MH * MFFN;
    const float* fc2b_l = fc2b + (size_t)l * MH;

    // freq_layer
    layernorm_kernel<0><<<MT, 256, 0, stream>>>(Xact, lng_l, lnb_l, MH, XN, nullptr);
    rfft_kernel<<<MB * MH, 256, 0, stream>>>(XN, FREQ);
    score_kernel<<<MB * MNH, 256, 0, stream>>>(FREQ, fi_l, SCOR);
    topk_kernel<<<MB * MNH, 256, 0, stream>>>(SCOR, MASKB);
    irfft_kernel<<<MB * MH, 256, 0, stream>>>(FREQ, MASKB, hw_l, BF1);
    cvt(ow_l, WBUF, (size_t)MH * MH);
    gemm(EP_F32_RES, BF1, WBUF, ob_l, Xact, Xact, MT, MH, MH);

    // ffn
    layernorm_kernel<1><<<MT, 256, 0, stream>>>(Xact, fng_l, fnb_l, MH, nullptr, BF1);
    cvt(fc1w_l, WBUF, (size_t)MFFN * MH);
    gemm(EP_BF16_GELU, BF1, WBUF, fc1b_l, nullptr, HBF, MT, MFFN, MH);
    cvt(fc2w_l, WBUF, (size_t)MH * MFFN);
    gemm(EP_F32_RES, HBF, WBUF, fc2b_l, Xact, Xact, MT, MH, MFFN);
  }

  // ---- epilogue: final LN, output projection, logits vs emb^T ----
  layernorm_kernel<1><<<MT, 256, 0, stream>>>(Xact, ong, onb, MH, nullptr, BF1);
  cvt(opw, WBUF, (size_t)ME * MH);
  gemm(EP_BF16, BF1, WBUF, opb, nullptr, BF2, MT, ME, MH);
  gemm(EP_F32, BF2, EMBB, nullptr, nullptr, out, MT, MV, ME);
}
